// MoELayer_4964982194281
// MI455X (gfx1250) — matrix operand order repaired and checked by reading, not compiled, not run on hardware
//
#include <hip/hip_runtime.h>
#include <hip/hip_bf16.h>

// ---------------------------------------------------------------------------
// MoE top-2 SwiGLU FFN for MI455X (gfx1250, wave32, WMMA bf16).
//   B=2 S=2048 -> T=4096 tokens, H=1024, I=2816, E=8, top-2 routing.
// Pipeline (all on `stream`):
//   1. memset counts + out
//   2. k_conv_x  : x f32 -> bf16 row-major [T,H]
//   3. k_conv_w  : Wg/Wu/Wd f32 -> bf16 pre-swizzled into WMMA B-fragment
//                  layout (32x16 frags, lane L owns 32 contiguous bytes)
//   4. k_router  : logits, top-2, softmax, atomic append to per-expert lists
//   5. k_ffn     : per (expert, 16-token tile): g/u/down GEMMs with
//                  v_wmma_f32_16x16x32_bf16, f32 accum, fast-sigmoid SiLU in
//                  C-layout (v_rcp_f32 instead of IEEE divide), weighted
//                  atomic scatter-add to out (2 adds/elem -> commutative ->
//                  bitwise deterministic).
// ---------------------------------------------------------------------------

#define MOE_T 4096
#define MOE_H 1024
#define MOE_I 2816
#define MOE_E 8

typedef __attribute__((ext_vector_type(16))) __bf16 v16bf;
typedef __attribute__((ext_vector_type(8)))  float  v8f;

union FragAB { v16bf v; uint4 q[2]; };

__device__ __forceinline__ unsigned short f2bf(float f) {
  // round-to-nearest-even f32 -> bf16
  unsigned int u = __float_as_uint(f);
  u += 0x7fffu + ((u >> 16) & 1u);
  return (unsigned short)(u >> 16);
}

__device__ __forceinline__ v8f zero8() {
  v8f z;
#pragma unroll
  for (int i = 0; i < 8; ++i) z[i] = 0.f;
  return z;
}

// ------------------------------- x -> bf16 ---------------------------------
__global__ __launch_bounds__(256) void k_conv_x(const float* __restrict__ x,
                                                unsigned short* __restrict__ xb,
                                                int n4) {
  int i = blockIdx.x * 256 + threadIdx.x;
  if (i >= n4) return;
  float4 f = ((const float4*)x)[i];
  uint2 o;
  o.x = (unsigned)f2bf(f.x) | ((unsigned)f2bf(f.y) << 16);
  o.y = (unsigned)f2bf(f.z) | ((unsigned)f2bf(f.w) << 16);
  ((uint2*)xb)[i] = o;
}

// ------------------- W [E][R][K] f32 -> bf16 B-fragments -------------------
// Fragment fi = nt*(K/32)+ks holds B (32x16): B[k,n] = W[nt*16+n][ks*32+k].
// ISA B layout (16-bit, 32x16): lane L (n=L%16): lanes<16 hold K=0..7,16..23,
// lanes>=16 hold K=8..15,24..31, K ascending in pairs across the 8 VGPRs
// => lane L's registers == W[row][k0+(L/16)*8 .. +8] then W[row][k0+16+(L/16)*8 .. +8].
__global__ __launch_bounds__(256) void k_conv_w(const float* __restrict__ src,
                                                unsigned short* __restrict__ dst,
                                                int R, int K, int total) {
  int gid = blockIdx.x * 256 + threadIdx.x;
  if (gid >= total) return;
  int perE = (R * K) >> 3;          // 8-element groups per expert matrix
  int e    = gid / perE;
  int lg   = gid - e * perE;
  int gpr  = K >> 3;                // groups per row
  int row  = lg / gpr;
  int col0 = (lg - row * gpr) << 3; // first of 8 consecutive k's
  int nt = row >> 4, n = row & 15;
  int ks = col0 >> 5, rem = col0 & 31;     // rem in {0,8,16,24}
  int lane = (((rem >> 3) & 1) << 4) + n;  // which lane's 32B run
  int slot = rem >> 4;                     // first/second 16B of the run
  size_t mbase = (size_t)e * (size_t)R * (size_t)K;
  size_t fi    = (size_t)nt * (size_t)(K >> 5) + (size_t)ks;
  size_t doff  = mbase + fi * 512 + (size_t)lane * 16 + (size_t)slot * 8;
  const float* s = src + mbase + (size_t)row * K + col0;
  float4 f0 = ((const float4*)s)[0];
  float4 f1 = ((const float4*)s)[1];
  uint4 o;
  o.x = (unsigned)f2bf(f0.x) | ((unsigned)f2bf(f0.y) << 16);
  o.y = (unsigned)f2bf(f0.z) | ((unsigned)f2bf(f0.w) << 16);
  o.z = (unsigned)f2bf(f1.x) | ((unsigned)f2bf(f1.y) << 16);
  o.w = (unsigned)f2bf(f1.z) | ((unsigned)f2bf(f1.w) << 16);
  *(uint4*)(dst + doff) = o;
}

// --------------------------------- router ----------------------------------
__global__ __launch_bounds__(256) void k_router(const float* __restrict__ x,
                                                const float* __restrict__ rw,
                                                int* __restrict__ counts,
                                                int* __restrict__ toks,
                                                float* __restrict__ wts) {
  __shared__ float s_rw[MOE_E * MOE_H];     // 32 KB
  int tid = threadIdx.x;
  for (int i = tid; i < MOE_E * MOE_H; i += 256) s_rw[i] = rw[i];
  __syncthreads();
  int lane = tid & 31, w = tid >> 5;
  int t = blockIdx.x * 8 + w;               // one wave per token
  const float* xr = x + (size_t)t * MOE_H;
  float acc[MOE_E];
#pragma unroll
  for (int e = 0; e < MOE_E; ++e) acc[e] = 0.f;
  for (int k = lane; k < MOE_H; k += 32) {
    float xv = xr[k];
#pragma unroll
    for (int e = 0; e < MOE_E; ++e) acc[e] += xv * s_rw[e * MOE_H + k];
  }
#pragma unroll
  for (int e = 0; e < MOE_E; ++e)
    for (int off = 16; off > 0; off >>= 1)
      acc[e] += __shfl_xor(acc[e], off, 32);
  if (lane == 0) {
    float v1 = -3.4e38f, v2 = -3.4e38f; int i1 = 0, i2 = 0;
#pragma unroll
    for (int e = 0; e < MOE_E; ++e) {       // strict '>' keeps lower index on ties
      float v = acc[e];
      if (v > v1)      { v2 = v1; i2 = i1; v1 = v; i1 = e; }
      else if (v > v2) { v2 = v;  i2 = e; }
    }
    float d  = __expf(v2 - v1);             // <= 1, stable
    float w1 = 1.f / (1.f + d);             // exact divide: routing weights
    float w2 = d   / (1.f + d);             // feed the final combine
    int s1 = atomicAdd(&counts[i1], 1);
    toks[i1 * MOE_T + s1] = t; wts[i1 * MOE_T + s1] = w1;
    int s2 = atomicAdd(&counts[i2], 1);
    toks[i2 * MOE_T + s2] = t; wts[i2 * MOE_T + s2] = w2;
  }
}

// ---------------------------------- FFN ------------------------------------
// Grid: E * (T/16) blocks, 256 threads (8 waves). Block = (expert e, 16-token
// tile). Wave w: up-proj columns [w*16, w*16+16) of each 128-wide I-chunk,
// down-proj output n-tiles [w*8, w*8+8) of H/16=64.
__global__ __launch_bounds__(256) void k_ffn(const unsigned short* __restrict__ xb,
                                             const unsigned short* __restrict__ wgF,
                                             const unsigned short* __restrict__ wuF,
                                             const unsigned short* __restrict__ wdF,
                                             const int* __restrict__ counts,
                                             const int* __restrict__ toks,
                                             const float* __restrict__ wts,
                                             float* __restrict__ out) {
  int e    = blockIdx.x >> 8;          // 256 tiles per expert
  int tile = blockIdx.x & 255;
  int cnt  = counts[e];
  if (tile * 16 >= cnt) return;        // block-uniform exit
  int nvalid = min(16, cnt - tile * 16);

  __shared__ unsigned short lx[16 * 1032];  // x tile, rows padded +8 bf16
  __shared__ unsigned short lh[16 * 136];   // h tile (16x128), rows padded +8
  __shared__ int   s_tok[16];
  __shared__ float s_w[16];

  int tid = threadIdx.x, lane = tid & 31, w = tid >> 5;
  int mrow = lane & 15, khalf = lane >> 4;

  if (tid < 16) {
    bool valid = tid < nvalid;
    s_tok[tid] = valid ? toks[e * MOE_T + tile * 16 + tid] : 0;
    s_w[tid]   = valid ? wts [e * MOE_T + tile * 16 + tid] : 0.f;
  }
  __syncthreads();
  // gather 16 token rows (2 KB each) into LDS, 256 threads x 8B, coalesced
  for (int r = 0; r < 16; ++r) {
    const uint2* s = (const uint2*)(xb + (size_t)s_tok[r] * MOE_H);
    ((uint2*)(&lx[r * 1032]))[tid] = s[tid];
  }
  __syncthreads();

  v8f accd[8];
#pragma unroll
  for (int j = 0; j < 8; ++j) accd[j] = zero8();

  const size_t wbase = (size_t)e * MOE_I * MOE_H;

  for (int c = 0; c < 22; ++c) {            // 22 chunks of 128 over I=2816
    v8f cg = zero8(), cu = zero8();
    const unsigned short* gB = wgF + wbase + ((size_t)(c * 8 + w) * 32) * 512;
    const unsigned short* uB = wuF + wbase + ((size_t)(c * 8 + w) * 32) * 512;
#pragma unroll 4
    for (int ks = 0; ks < 32; ++ks) {       // K = 1024 = 32 x 32
      FragAB a, bg, bu;
      // A (x) from LDS per ISA A-layout: lane<16 K=[0..7]+[16..23], lane>=16 +8
      const unsigned short* xr = &lx[mrow * 1032 + ks * 32 + khalf * 8];
      a.q[0] = *(const uint4*)xr;
      a.q[1] = *(const uint4*)(xr + 16);
      // B pre-swizzled: lane L's 32 contiguous bytes
      const uint4* gq = (const uint4*)(gB + (size_t)ks * 512 + lane * 16);
      bg.q[0] = gq[0]; bg.q[1] = gq[1];
      const uint4* uq = (const uint4*)(uB + (size_t)ks * 512 + lane * 16);
      bu.q[0] = uq[0]; bu.q[1] = uq[1];
      cg = __builtin_amdgcn_wmma_f32_16x16x32_bf16(false, a.v, false, bg.v,
                                                   (short)0, cg, false, false);
      cu = __builtin_amdgcn_wmma_f32_16x16x32_bf16(false, a.v, false, bu.v,
                                                   (short)0, cu, false, false);
    }
    __syncthreads();   // previous chunk's lh fully consumed
    // h = silu(g) * u in C-layout (VGPR r -> M = r, +8 if lane>=16).
    // Fast sigmoid: v_rcp_f32 (1 ulp) instead of IEEE divide — h is rounded
    // to bf16 immediately, so rcp accuracy is more than sufficient.
#pragma unroll
    for (int r = 0; r < 8; ++r) {
      float g = cg[r], u = cu[r];
      float sig = __builtin_amdgcn_rcpf(1.f + __expf(-g));
      float hv  = g * sig * u;
      int row = khalf ? (r + 8) : r;
      lh[row * 136 + w * 16 + mrow] = f2bf(hv);
    }
    __syncthreads();
    // down-proj: y[16,1024] += h[16,128] @ Wd_chunk^T, K=128 -> 4 k-steps
#pragma unroll
    for (int ks2 = 0; ks2 < 4; ++ks2) {
      FragAB a;
      const unsigned short* hr = &lh[mrow * 136 + ks2 * 32 + khalf * 8];
      a.q[0] = *(const uint4*)hr;
      a.q[1] = *(const uint4*)(hr + 16);
#pragma unroll
      for (int j = 0; j < 8; ++j) {
        FragAB b;
        const uint4* bq = (const uint4*)(wdF + (size_t)e * MOE_H * MOE_I +
            ((size_t)(w * 8 + j) * 88 + (size_t)(c * 4 + ks2)) * 512 + lane * 16);
        b.q[0] = bq[0]; b.q[1] = bq[1];
        accd[j] = __builtin_amdgcn_wmma_f32_16x16x32_bf16(false, a.v, false, b.v,
                                                (short)0, accd[j], false, false);
      }
    }
  }
  // epilogue: weighted scatter-add; exactly 2 atomic adds per out element
#pragma unroll
  for (int j = 0; j < 8; ++j) {
    int col = (w * 8 + j) * 16 + mrow;
#pragma unroll
    for (int r = 0; r < 8; ++r) {
      int row = khalf ? (r + 8) : r;
      if (row < nvalid)
        atomicAdd(out + (size_t)s_tok[row] * MOE_H + col, s_w[row] * accd[j][r]);
    }
  }
}

// ------------------------------ host launch --------------------------------
extern "C" void kernel_launch(void* const* d_in, const int* in_sizes, int n_in,
                              void* d_out, int out_size, void* d_ws, size_t ws_size,
                              hipStream_t stream) {
  (void)in_sizes; (void)n_in;
  const float* x  = (const float*)d_in[0];
  const float* rw = (const float*)d_in[1];
  const float* Wg = (const float*)d_in[2];
  const float* Wu = (const float*)d_in[3];
  const float* Wd = (const float*)d_in[4];
  float* out = (float*)d_out;

  char* ws = (char*)d_ws;
  size_t off = 0;
  auto take = [&](size_t bytes) -> char* {
    char* p = ws + off;
    off += (bytes + 255) & ~(size_t)255;
    return p;
  };
  int*            counts = (int*)  take(MOE_E * sizeof(int));
  int*            toks   = (int*)  take((size_t)MOE_E * MOE_T * sizeof(int));
  float*          wts    = (float*)take((size_t)MOE_E * MOE_T * sizeof(float));
  unsigned short* xb     = (unsigned short*)take((size_t)MOE_T * MOE_H * 2);
  unsigned short* wgF    = (unsigned short*)take((size_t)MOE_E * MOE_I * MOE_H * 2);
  unsigned short* wuF    = (unsigned short*)take((size_t)MOE_E * MOE_I * MOE_H * 2);
  unsigned short* wdF    = (unsigned short*)take((size_t)MOE_E * MOE_H * MOE_I * 2);

  if (off > ws_size) {  // workspace too small: emit deterministic zeros
    hipMemsetAsync(d_out, 0, (size_t)out_size * sizeof(float), stream);
    return;
  }

  hipMemsetAsync(counts, 0, MOE_E * sizeof(int), stream);
  hipMemsetAsync(d_out, 0, (size_t)out_size * sizeof(float), stream);

  {
    int n4 = MOE_T * MOE_H / 4;
    k_conv_x<<<(n4 + 255) / 256, 256, 0, stream>>>(x, xb, n4);
  }
  {
    int tot = MOE_E * MOE_I * MOE_H / 8;     // same element count for all three
    int blocks = (tot + 255) / 256;
    k_conv_w<<<blocks, 256, 0, stream>>>(Wg, wgF, MOE_I, MOE_H, tot);
    k_conv_w<<<blocks, 256, 0, stream>>>(Wu, wuF, MOE_I, MOE_H, tot);
    k_conv_w<<<blocks, 256, 0, stream>>>(Wd, wdF, MOE_H, MOE_I, tot);
  }
  k_router<<<MOE_T / 8, 256, 0, stream>>>(x, rw, counts, toks, wts);
  k_ffn<<<MOE_E * (MOE_T / 16), 256, 0, stream>>>(xb, wgF, wuF, wdF,
                                                  counts, toks, wts, out);
}
